// CorticalColumnGroup_2491081031793
// MI455X (gfx1250) — compile-verified, matrix operand order checked by reading, not run on hardware
//
#include <hip/hip_runtime.h>
#include <hip/hip_bf16.h>

typedef __attribute__((ext_vector_type(16))) _Float16 v16h;
typedef __attribute__((ext_vector_type(8)))  _Float16 v8h;
typedef __attribute__((ext_vector_type(8)))  float    v8f;

#define BSZ   8
#define NTOK  512
#define NGRP  64
#define DCOL  128
#define DATT  64
#define DHID  512
#define LN_EPS 1e-5f

// ---------------- WMMA fragment helpers (CDNA5 16x16x32 f16 layouts) -----------
// All loaders fetch two contiguous 16-byte chunks per lane.

__device__ __forceinline__ v8f wmma(v16h a, v16h b, v8f c) {
  return __builtin_amdgcn_wmma_f32_16x16x32_f16(false, a, false, b, (short)0, c,
                                                false, false);
}

// A fragment 16x32 from row-major A[m][k]; lda in elements (multiple of 8).
__device__ __forceinline__ v16h frag_a(const _Float16* A, int lda) {
  int lane = threadIdx.x & 31;
  int row  = lane & 15;
  int kh   = (lane >> 4) * 8;
  v8h lo = *(const v8h*)(A + (size_t)row * lda + kh);
  v8h hi = *(const v8h*)(A + (size_t)row * lda + 16 + kh);
  v16h r;
#pragma unroll
  for (int i = 0; i < 8; ++i) { r[i] = lo[i]; r[8 + i] = hi[i]; }
  return r;
}

// B fragment 32x16 where the source is the TRANSPOSE of B: Bt[n][k] row-major.
// (i.e. B[k][n] = Bt[n][k]); contiguous 16B reads per lane.
__device__ __forceinline__ v16h frag_bt(const _Float16* Bt, int ldt) {
  int lane = threadIdx.x & 31;
  int col  = lane & 15;
  int kh   = (lane >> 4) * 8;
  v8h lo = *(const v8h*)(Bt + (size_t)col * ldt + kh);
  v8h hi = *(const v8h*)(Bt + (size_t)col * ldt + 16 + kh);
  v16h r;
#pragma unroll
  for (int i = 0; i < 8; ++i) { r[i] = lo[i]; r[8 + i] = hi[i]; }
  return r;
}

// 256 threads: LayerNorm a 16x128 tile of x into f16 LDS buffer sA[tok][128].
__device__ __forceinline__ void ln_tile_to_lds(const float* __restrict__ xrow0,
                                               int row_stride,
                                               const float* __restrict__ gamma,
                                               const float* __restrict__ beta,
                                               _Float16* sA) {
  int t = threadIdx.x;
  int row = t >> 4, sub = t & 15;
  const float* xr = xrow0 + (size_t)row * row_stride;
  float v[8];
  float s1 = 0.f, s2 = 0.f;
#pragma unroll
  for (int i = 0; i < 8; ++i) {
    float f = xr[sub * 8 + i];
    v[i] = f; s1 += f; s2 += f * f;
  }
#pragma unroll
  for (int m = 1; m < 16; m <<= 1) {
    s1 += __shfl_xor(s1, m, 32);
    s2 += __shfl_xor(s2, m, 32);
  }
  float mu  = s1 * (1.f / 128.f);
  float var = s2 * (1.f / 128.f) - mu * mu;
  float inv = rsqrtf(var + LN_EPS);
#pragma unroll
  for (int i = 0; i < 8; ++i) {
    int c = sub * 8 + i;
    sA[row * 128 + c] = (_Float16)((v[i] - mu) * inv * gamma[c] + beta[c]);
  }
}

// 256 threads: LayerNorm a 32x128 tile (8 lanes per row, 16 elems each).
__device__ __forceinline__ void ln_tile32_to_lds(const float* __restrict__ xrow0,
                                                 int row_stride,
                                                 const float* __restrict__ gamma,
                                                 const float* __restrict__ beta,
                                                 _Float16* sA) {
  int t = threadIdx.x;
  int row = t >> 3, sub = t & 7;
  const float* xr = xrow0 + (size_t)row * row_stride + sub * 16;
  float v[16];
  float s1 = 0.f, s2 = 0.f;
#pragma unroll
  for (int i = 0; i < 16; ++i) {
    float f = xr[i];
    v[i] = f; s1 += f; s2 += f * f;
  }
#pragma unroll
  for (int m = 1; m < 8; m <<= 1) {
    s1 += __shfl_xor(s1, m, 32);
    s2 += __shfl_xor(s2, m, 32);
  }
  float mu  = s1 * (1.f / 128.f);
  float var = s2 * (1.f / 128.f) - mu * mu;
  float inv = rsqrtf(var + LN_EPS);
#pragma unroll
  for (int i = 0; i < 16; ++i) {
    int c = sub * 16 + i;
    sA[row * 128 + c] = (_Float16)((v[i] - mu) * inv * gamma[c] + beta[c]);
  }
}

// ---------------- conversion kernels -------------------------------------------

// Transpose+convert: src[g][K][Nc] f32 -> dst[g][Nc][K] f16.  K is a power of 2.
__global__ void __launch_bounds__(256)
f32_to_f16_t_kernel(const float* __restrict__ src, _Float16* __restrict__ dst,
                    int K, int Nc, int kshift) {
  int g = blockIdx.y;
  size_t base = (size_t)g * K * Nc;
  int idx = blockIdx.x * 256 + threadIdx.x;
  if (idx >= K * Nc) return;
  int k = idx & (K - 1);
  int n = idx >> kshift;
  dst[base + (size_t)n * K + k] = (_Float16)src[base + (size_t)k * Nc + n];
}

__global__ void __launch_bounds__(512)
zero_kernel(float* __restrict__ p, size_t n) {
  size_t i = (size_t)blockIdx.x * blockDim.x + threadIdx.x;
  if (i < n) p[i] = 0.f;
}

// ---------------- elementwise mixes --------------------------------------------

__global__ void __launch_bounds__(256)
mix_kernel(const float* __restrict__ xin, const float* __restrict__ mixm,
           float* __restrict__ xout) {
  __shared__ float sm[256];
  sm[threadIdx.x] = mixm[threadIdx.x];
  __syncthreads();
  size_t idx = (size_t)blockIdx.x * blockDim.x + threadIdx.x;
  int d  = (int)(idx & 127);
  int xb = (int)((idx >> 7) & 3);
  int r  = (int)(idx >> 9);
  const float* base = xin + ((size_t)r * NGRP + xb * 16) * DCOL + d;
  float v[16];
#pragma unroll
  for (int c = 0; c < 16; ++c) v[c] = base[(size_t)c * DCOL];
  float* ob = xout + ((size_t)r * NGRP + xb * 16) * DCOL + d;
#pragma unroll
  for (int e = 0; e < 16; ++e) {
    float s = v[e];
#pragma unroll
    for (int c = 0; c < 16; ++c) s += sm[e * 16 + c] * v[c];
    ob[(size_t)e * DCOL] = s;
  }
}

__global__ void __launch_bounds__(256)
cross_kernel(float* __restrict__ x, const float* __restrict__ crs) {
  size_t idx = (size_t)blockIdx.x * blockDim.x + threadIdx.x;
  int d = (int)(idx & 127);
  int c = (int)((idx >> 7) & 15);
  int r = (int)(idx >> 11);
  size_t base = ((size_t)r * NGRP + c) * DCOL + d;
  float v[4];
#pragma unroll
  for (int xb = 0; xb < 4; ++xb) v[xb] = x[base + (size_t)xb * 16 * DCOL];
#pragma unroll
  for (int f = 0; f < 4; ++f) {
    float s = v[f];
#pragma unroll
    for (int xb = 0; xb < 4; ++xb) s += crs[f * 4 + xb] * v[xb];
    x[base + (size_t)f * 16 * DCOL] = s;
  }
}

// ---------------- QKV projection (LN fused, transposed compute) ----------------
// out^T[e][tok] = W^T[e][:] x LN(x)^T[:][tok].
// Q/K stored [bg][n][dk] (contiguous v8h stores); V stored transposed VT[bg][d][n].
__global__ void __launch_bounds__(256)
qkv_kernel(const float* __restrict__ x,
           const float* __restrict__ ng, const float* __restrict__ nb,
           const _Float16* __restrict__ qwT, const float* __restrict__ qb,
           const _Float16* __restrict__ kwT, const float* __restrict__ kb,
           const _Float16* __restrict__ vwT, const float* __restrict__ vb,
           _Float16* __restrict__ Qo, _Float16* __restrict__ Ko,
           _Float16* __restrict__ VT) {
  __shared__ _Float16 sA[16 * 128];
  int tile = blockIdx.x, g = blockIdx.y;
  int r0 = tile * 16;
  int bidx = r0 >> 9, n0 = r0 & 511;
  __builtin_prefetch(qwT + (size_t)g * 64 * 128, 0, 1);
  __builtin_prefetch(vwT + (size_t)g * 128 * 128, 0, 1);
  ln_tile_to_lds(x + (size_t)r0 * (NGRP * DCOL) + (size_t)g * DCOL, NGRP * DCOL,
                 ng + g * DCOL, nb + g * DCOL, sA);
  __syncthreads();
  int w = threadIdx.x >> 5;
  int lane = threadIdx.x & 31;
  int col = lane & 15, rb = (lane >> 4) * 8;
#pragma unroll
  for (int j = 0; j < 2; ++j) {
    int ti = w * 2 + j;  // 0..3 Q, 4..7 K, 8..15 V
    const _Float16* WT; const float* bias; int e0;
    bool isV = (ti >= 8);
    if (ti < 4)      { WT = qwT + (size_t)g * 64 * 128;  bias = qb + g * 64;  e0 = ti * 16; }
    else if (ti < 8) { WT = kwT + (size_t)g * 64 * 128;  bias = kb + g * 64;  e0 = (ti - 4) * 16; }
    else             { WT = vwT + (size_t)g * 128 * 128; bias = vb + g * 128; e0 = (ti - 8) * 16; }
    v8f acc = {};
#pragma unroll
    for (int kk = 0; kk < 4; ++kk) {
      v16h a = frag_a(WT + (size_t)e0 * 128 + kk * 32, 128);
      v16h b = frag_bt(sA + kk * 32, 128);
      acc = wmma(a, b, acc);
    }
    if (!isV) {
      _Float16* Out = (ti < 4) ? Qo : Ko;
      v8h pack;
#pragma unroll
      for (int i = 0; i < 8; ++i) pack[i] = (_Float16)(acc[i] + bias[e0 + rb + i]);
      *(v8h*)(Out + ((size_t)(bidx * NGRP + g) * NTOK + n0 + col) * DATT + e0 + rb) = pack;
    } else {
#pragma unroll
      for (int i = 0; i < 8; ++i)
        VT[((size_t)(bidx * NGRP + g) * DCOL + e0 + rb + i) * NTOK + n0 + col] =
            (_Float16)(acc[i] + bias[e0 + rb + i]);
    }
  }
}

// ---------------- attention core -----------------------------------------------
// grid (32 query tiles, 512 bg).  S = Q K^T (contiguous frags), softmax in LDS,
// then AO^T[d][q] = V^T P^T (all-contiguous frags); AO stored [bg][n][128].
__global__ void __launch_bounds__(256)
attn_kernel(const _Float16* __restrict__ Q, const _Float16* __restrict__ K,
            const _Float16* __restrict__ VT, _Float16* __restrict__ AO) {
  __shared__ _Float16 sQ[16 * 64];
  __shared__ float sS[16 * 512];
  int qt = blockIdx.x, bg = blockIdx.y;
  const _Float16* Qg  = Q  + (size_t)bg * NTOK * DATT;
  const _Float16* Kg  = K  + (size_t)bg * NTOK * DATT;
  const _Float16* VTg = VT + (size_t)bg * DCOL * NTOK;
  int q0 = qt * 16;
  __builtin_prefetch(Kg, 0, 1);
  __builtin_prefetch(VTg, 0, 1);
  {
    int t = threadIdx.x;  // 1024 halves total
#pragma unroll
    for (int i = 0; i < 4; ++i) sQ[t * 4 + i] = Qg[(size_t)q0 * DATT + t * 4 + i];
  }
  __syncthreads();
  int w = threadIdx.x >> 5;
  int lane = threadIdx.x & 31;
  int col = lane & 15, rb = (lane >> 4) * 8;
#pragma unroll
  for (int ct = 0; ct < 4; ++ct) {
    int k0 = w * 64 + ct * 16;
    v8f acc = {};
#pragma unroll
    for (int kk = 0; kk < 2; ++kk) {
      v16h a = frag_a(sQ + kk * 32, 64);
      v16h b = frag_bt(Kg + (size_t)k0 * DATT + kk * 32, 64);  // B = K^T
      acc = wmma(a, b, acc);
    }
#pragma unroll
    for (int i = 0; i < 8; ++i)
      sS[(rb + i) * 512 + k0 + col] = acc[i] * 0.125f;  // 1/sqrt(64)
  }
  __syncthreads();
  _Float16* sP = (_Float16*)sS;  // aliases sS; reads staged to regs first
  {
    int row = threadIdx.x >> 4, sub = threadIdx.x & 15;
    float e[32];
    float m = -3.4e38f;
#pragma unroll
    for (int i = 0; i < 32; ++i) {
      e[i] = sS[row * 512 + sub * 32 + i];
      m = fmaxf(m, e[i]);
    }
#pragma unroll
    for (int msk = 1; msk < 16; msk <<= 1) m = fmaxf(m, __shfl_xor(m, msk, 32));
    float sum = 0.f;
#pragma unroll
    for (int i = 0; i < 32; ++i) { e[i] = __expf(e[i] - m); sum += e[i]; }
#pragma unroll
    for (int msk = 1; msk < 16; msk <<= 1) sum += __shfl_xor(sum, msk, 32);
    float rinv = 1.f / sum;
    __syncthreads();  // everyone done reading f32 scores
#pragma unroll
    for (int i = 0; i < 32; ++i)
      sP[row * 512 + sub * 32 + i] = (_Float16)(e[i] * rinv);
  }
  __syncthreads();
  {
    int d0 = w * 16;
    v8f acc = {};
#pragma unroll
    for (int kk = 0; kk < 16; ++kk) {
      v16h a = frag_a(VTg + (size_t)d0 * NTOK + kk * 32, NTOK);  // rows = d
      v16h b = frag_bt(sP + kk * 32, 512);                       // B = P^T
      acc = wmma(a, b, acc);
    }
    v8h pack;
#pragma unroll
    for (int i = 0; i < 8; ++i) pack[i] = (_Float16)acc[i];
    *(v8h*)(AO + ((size_t)bg * NTOK + q0 + col) * DCOL + d0 + rb) = pack;
  }
}

// ---------------- O-projection + residual (transposed compute) -----------------
__global__ void __launch_bounds__(256)
oproj_kernel(float* __restrict__ x, const _Float16* __restrict__ AO,
             const _Float16* __restrict__ owT, const float* __restrict__ ob) {
  __shared__ _Float16 sAO[16 * 128];
  int tile = blockIdx.x, g = blockIdx.y;
  int r0 = tile * 16;
  int bidx = r0 >> 9, n0 = r0 & 511;
  __builtin_prefetch(owT + (size_t)g * 128 * 128, 0, 1);
  {
    int t = threadIdx.x;
    int row = t >> 4, sub = t & 15;
    const _Float16* src = AO + ((size_t)(bidx * NGRP + g) * NTOK + n0 + row) * DCOL;
    *(v8h*)(sAO + row * 128 + sub * 8) = *(const v8h*)(src + sub * 8);
  }
  __syncthreads();
  int w = threadIdx.x >> 5;
  int lane = threadIdx.x & 31;
  int col = lane & 15, rb = (lane >> 4) * 8;
  int e0 = w * 16;
  const _Float16* WT = owT + (size_t)g * 128 * 128;
  v8f acc = {};
#pragma unroll
  for (int kk = 0; kk < 4; ++kk) {
    v16h a = frag_a(WT + (size_t)e0 * 128 + kk * 32, 128);
    v16h b = frag_bt(sAO + kk * 32, 128);
    acc = wmma(a, b, acc);
  }
  float* xp = x + ((size_t)(r0 + col) * NGRP + g) * DCOL + e0 + rb;
#pragma unroll
  for (int i = 0; i < 8; ++i) xp[i] += acc[i] + ob[g * 128 + e0 + rb + i];
}

// ---------------- FFN layer: x += down(gelu(up(LN(x)))), 32-token tiles --------
// Each weight A-fragment is reused by two token column-tiles (better WMMA:load).
__global__ void __launch_bounds__(256)
ffn_kernel(float* __restrict__ x,
           const float* __restrict__ ng, const float* __restrict__ nb,
           const _Float16* __restrict__ uwT, const float* __restrict__ ub,
           const _Float16* __restrict__ dwT, const float* __restrict__ db) {
  __shared__ _Float16 sA[32 * 128];
  __shared__ _Float16 sH[32 * 512];   // [tok][h]
  int tile = blockIdx.x, g = blockIdx.y;
  int r0 = tile * 32;
  const _Float16* WuT = uwT + (size_t)g * 512 * 128;  // [h][d]
  const _Float16* WdT = dwT + (size_t)g * 128 * 512;  // [e][h]
  __builtin_prefetch(WuT, 0, 1);
  __builtin_prefetch(WdT, 0, 1);
  ln_tile32_to_lds(x + (size_t)r0 * (NGRP * DCOL) + (size_t)g * DCOL, NGRP * DCOL,
                   ng + g * DCOL, nb + g * DCOL, sA);
  __syncthreads();
  int w = threadIdx.x >> 5;
  int lane = threadIdx.x & 31;
  int col = lane & 15, rb = (lane >> 4) * 8;
#pragma unroll
  for (int ct = 0; ct < 4; ++ct) {
    int h0 = w * 64 + ct * 16;
    v8f acc0 = {}, acc1 = {};
#pragma unroll
    for (int kk = 0; kk < 4; ++kk) {
      v16h a  = frag_a(WuT + (size_t)h0 * 128 + kk * 32, 128);
      v16h b0 = frag_bt(sA + kk * 32, 128);             // tokens 0..15
      v16h b1 = frag_bt(sA + 16 * 128 + kk * 32, 128);  // tokens 16..31
      acc0 = wmma(a, b0, acc0);
      acc1 = wmma(a, b1, acc1);
    }
    v8h p0, p1;
#pragma unroll
    for (int i = 0; i < 8; ++i) {
      float bv = ub[g * 512 + h0 + rb + i];
      float h0v = acc0[i] + bv;
      float h1v = acc1[i] + bv;
      p0[i] = (_Float16)(0.5f * h0v * (1.0f + erff(h0v * 0.70710678118654752f)));
      p1[i] = (_Float16)(0.5f * h1v * (1.0f + erff(h1v * 0.70710678118654752f)));
    }
    *(v8h*)(sH + (size_t)col * 512 + h0 + rb) = p0;
    *(v8h*)(sH + (size_t)(16 + col) * 512 + h0 + rb) = p1;
  }
  __syncthreads();
  int e0 = w * 16;
  v8f acc0 = {}, acc1 = {};
#pragma unroll
  for (int kk = 0; kk < 16; ++kk) {
    v16h a  = frag_a(WdT + (size_t)e0 * 512 + kk * 32, 512);
    v16h b0 = frag_bt(sH + kk * 32, 512);
    v16h b1 = frag_bt(sH + (size_t)16 * 512 + kk * 32, 512);
    acc0 = wmma(a, b0, acc0);
    acc1 = wmma(a, b1, acc1);
  }
  float* xp0 = x + ((size_t)(r0 + col) * NGRP + g) * DCOL + e0 + rb;
  float* xp1 = x + ((size_t)(r0 + 16 + col) * NGRP + g) * DCOL + e0 + rb;
#pragma unroll
  for (int i = 0; i < 8; ++i) {
    float bv = db[g * 128 + e0 + rb + i];
    xp0[i] += acc0[i] + bv;
    xp1[i] += acc1[i] + bv;
  }
}

// ---------------- final projection (128 -> 513) + epilogues --------------------
__global__ void __launch_bounds__(256)
proj_kernel(const float* __restrict__ x, const _Float16* __restrict__ pwT,
            const float* __restrict__ pb,
            float* __restrict__ kc, float* __restrict__ vp,
            float* __restrict__ qn, float* __restrict__ vc,
            float* __restrict__ wn) {
  __shared__ _Float16 sA[16 * 128];
  __shared__ float sPr[16 * 513];   // [tok][e]
  int tile = blockIdx.x, g = blockIdx.y;
  int r0 = tile * 16;
  __builtin_prefetch(pwT + (size_t)g * 513 * 128, 0, 1);
  {
    int t = threadIdx.x;
    int row = t >> 4, sub = t & 15;
    const float* src = x + ((size_t)(r0 + row) * NGRP + g) * DCOL;
#pragma unroll
    for (int i = 0; i < 8; ++i)
      sA[row * 128 + sub * 8 + i] = (_Float16)src[sub * 8 + i];
  }
  __syncthreads();
  const _Float16* WT = pwT + (size_t)g * 513 * 128;  // [e][d]
  int w = threadIdx.x >> 5;
  int lane = threadIdx.x & 31;
  int col = lane & 15, rb = (lane >> 4) * 8;
#pragma unroll
  for (int ct = 0; ct < 4; ++ct) {
    int e0 = w * 64 + ct * 16;
    v8f acc = {};
#pragma unroll
    for (int kk = 0; kk < 4; ++kk) {
      v16h a = frag_a(WT + (size_t)e0 * 128 + kk * 32, 128);
      v16h b = frag_bt(sA + kk * 32, 128);
      acc = wmma(a, b, acc);
    }
#pragma unroll
    for (int i = 0; i < 8; ++i)
      sPr[col * 513 + e0 + rb + i] = acc[i] + pb[g * 513 + e0 + rb + i];
  }
  if (threadIdx.x < 16) {  // column 512 (novelty logit) via scalar dot
    int row = threadIdx.x;
    float s = pb[g * 513 + 512];
    const _Float16* wrow = WT + (size_t)512 * 128;
#pragma unroll
    for (int k = 0; k < 128; ++k)
      s += (float)sA[row * 128 + k] * (float)wrow[k];
    sPr[row * 513 + 512] = s;
  }
  __syncthreads();
  {
    int row = threadIdx.x >> 4, sub = threadIdx.x & 15;
    float n1 = 0.f, n2 = 0.f;
#pragma unroll
    for (int i = 0; i < 8; ++i) {
      float a = sPr[row * 513 + sub * 8 + i];       n1 += a * a;
      float b = sPr[row * 513 + 256 + sub * 8 + i]; n2 += b * b;
    }
#pragma unroll
    for (int m = 1; m < 16; m <<= 1) {
      n1 += __shfl_xor(n1, m, 32);
      n2 += __shfl_xor(n2, m, 32);
    }
    float i1 = 1.f / fmaxf(sqrtf(n1), 1e-12f);
    float i2 = 1.f / fmaxf(sqrtf(n2), 1e-12f);
    size_t ro = ((size_t)(r0 + row) * NGRP + g) * DCOL;
#pragma unroll
    for (int i = 0; i < 8; ++i) {
      int c = sub * 8 + i;
      kc[ro + c] = sPr[row * 513 + c] * i1;
      vp[ro + c] = sPr[row * 513 + 128 + c];
      qn[ro + c] = sPr[row * 513 + 256 + c] * i2;
      vc[ro + c] = sPr[row * 513 + 384 + c];
    }
    if (sub == 0) {
      float z = sPr[row * 513 + 512];
      wn[(size_t)(r0 + row) * NGRP + g] = 1.f / (1.f + __expf(-z));
    }
  }
}

// ---------------- host launch --------------------------------------------------

extern "C" void kernel_launch(void* const* d_in, const int* in_sizes, int n_in,
                              void* d_out, int out_size, void* d_ws, size_t ws_size,
                              hipStream_t stream) {
  (void)in_sizes; (void)n_in; (void)out_size; (void)ws_size;
  const float* x_col   = (const float*)d_in[0];
  const float* mixm    = (const float*)d_in[1];
  const float* crs     = (const float*)d_in[2];
  const float* a_ng    = (const float*)d_in[3];
  const float* a_nb    = (const float*)d_in[4];
  const float* a_qw    = (const float*)d_in[5];
  const float* a_qb    = (const float*)d_in[6];
  const float* a_kw    = (const float*)d_in[7];
  const float* a_kb    = (const float*)d_in[8];
  const float* a_vw    = (const float*)d_in[9];
  const float* a_vb    = (const float*)d_in[10];
  const float* a_ow    = (const float*)d_in[11];
  const float* a_ob    = (const float*)d_in[12];
  const float* pre_ng  = (const float*)d_in[13];
  const float* pre_nb  = (const float*)d_in[14];
  const float* pre_uw  = (const float*)d_in[15];
  const float* pre_ub  = (const float*)d_in[16];
  const float* pre_dw  = (const float*)d_in[17];
  const float* pre_db  = (const float*)d_in[18];
  const float* post_ng = (const float*)d_in[19];
  const float* post_nb = (const float*)d_in[20];
  const float* post_uw = (const float*)d_in[21];
  const float* post_ub = (const float*)d_in[22];
  const float* post_dw = (const float*)d_in[23];
  const float* post_db = (const float*)d_in[24];
  const float* p_pw    = (const float*)d_in[25];
  const float* p_pb    = (const float*)d_in[26];

  float* out = (float*)d_out;
  const size_t SZ_X = 33554432, SZ_S = 262144;
  float* out_x  = out;                    // working x buffer == x_out
  float* out_su = out + SZ_X;
  float* out_kc = out_su + SZ_S;
  float* out_vp = out_kc + SZ_X;
  float* out_gt = out_vp + SZ_X;
  float* out_qn = out_gt + SZ_S;
  float* out_vc = out_qn + SZ_X;
  float* out_wn = out_vc + SZ_X;

  char* ws = (char*)d_ws;
  size_t off = 0;
  auto walloc = [&](size_t elems) {
    void* p = ws + off;
    off += (elems * sizeof(_Float16) + 255) & ~(size_t)255;
    return (_Float16*)p;
  };
  _Float16* Q16  = walloc((size_t)BSZ * NTOK * NGRP * DATT);
  _Float16* K16  = walloc((size_t)BSZ * NTOK * NGRP * DATT);
  _Float16* VT16 = walloc((size_t)BSZ * NTOK * NGRP * DCOL);
  _Float16* AO16 = walloc((size_t)BSZ * NTOK * NGRP * DCOL);
  _Float16* qwT = walloc((size_t)NGRP * 64 * 128);
  _Float16* kwT = walloc((size_t)NGRP * 64 * 128);
  _Float16* vwT = walloc((size_t)NGRP * 128 * 128);
  _Float16* owT = walloc((size_t)NGRP * 128 * 128);
  _Float16* uwpreT  = walloc((size_t)2 * NGRP * 512 * 128);
  _Float16* dwpreT  = walloc((size_t)2 * NGRP * 128 * 512);
  _Float16* uwpostT = walloc((size_t)2 * NGRP * 512 * 128);
  _Float16* dwpostT = walloc((size_t)2 * NGRP * 128 * 512);
  _Float16* pwT     = walloc((size_t)NGRP * 513 * 128);

  // transpose-convert weights: src[g][K][Nc] -> dst[g][Nc][K]
  auto convT = [&](const float* s, _Float16* d, int groups, int K, int Nc, int ksh) {
    int per = K * Nc;
    f32_to_f16_t_kernel<<<dim3((per + 255) / 256, groups), 256, 0, stream>>>(
        s, d, K, Nc, ksh);
  };
  convT(a_qw, qwT, NGRP, 128, 64, 7);
  convT(a_kw, kwT, NGRP, 128, 64, 7);
  convT(a_vw, vwT, NGRP, 128, 128, 7);
  convT(a_ow, owT, NGRP, 128, 128, 7);
  convT(pre_uw,  uwpreT,  2 * NGRP, 128, 512, 7);
  convT(pre_dw,  dwpreT,  2 * NGRP, 512, 128, 9);
  convT(post_uw, uwpostT, 2 * NGRP, 128, 512, 7);
  convT(post_dw, dwpostT, 2 * NGRP, 512, 128, 9);
  convT(p_pw,    pwT,     NGRP, 128, 513, 7);

  // 1) channel mix into out_x (fully overwrites x region)
  mix_kernel<<<8192, 256, 0, stream>>>(x_col, mixm, out_x);
  // 2) attention
  qkv_kernel<<<dim3(256, 64), 256, 0, stream>>>(out_x, a_ng, a_nb,
                                                qwT, a_qb, kwT, a_kb,
                                                vwT, a_vb, Q16, K16, VT16);
  attn_kernel<<<dim3(32, 512), 256, 0, stream>>>(Q16, K16, VT16, AO16);
  oproj_kernel<<<dim3(256, 64), 256, 0, stream>>>(out_x, AO16, owT, a_ob);
  // 3) ffn_pre (2 layers)
  for (int l = 0; l < 2; ++l) {
    ffn_kernel<<<dim3(128, 64), 256, 0, stream>>>(
        out_x, pre_ng + (size_t)l * NGRP * 128, pre_nb + (size_t)l * NGRP * 128,
        uwpreT + (size_t)l * NGRP * 512 * 128, pre_ub + (size_t)l * NGRP * 512,
        dwpreT + (size_t)l * NGRP * 128 * 512, pre_db + (size_t)l * NGRP * 128);
  }
  // 4) cross mix (in place)
  cross_kernel<<<32768, 256, 0, stream>>>(out_x, crs);
  // 5) ffn_post (2 layers) -> out_x now equals x_out
  for (int l = 0; l < 2; ++l) {
    ffn_kernel<<<dim3(128, 64), 256, 0, stream>>>(
        out_x, post_ng + (size_t)l * NGRP * 128, post_nb + (size_t)l * NGRP * 128,
        uwpostT + (size_t)l * NGRP * 512 * 128, post_ub + (size_t)l * NGRP * 512,
        dwpostT + (size_t)l * NGRP * 128 * 512, post_db + (size_t)l * NGRP * 128);
  }
  // 6) projection + epilogues
  proj_kernel<<<dim3(256, 64), 256, 0, stream>>>(out_x, pwT, p_pb,
                                                 out_kc, out_vp, out_qn,
                                                 out_vc, out_wn);
  // 7) surprise / gate are exactly zero
  zero_kernel<<<(unsigned)((SZ_S + 511) / 512), 512, 0, stream>>>(out_su, SZ_S);
  zero_kernel<<<(unsigned)((SZ_S + 511) / 512), 512, 0, stream>>>(out_gt, SZ_S);
}
